// PN2_50440095924774
// MI455X (gfx1250) — matrix operand order repaired and checked by reading, not compiled, not run on hardware
//
#include <hip/hip_runtime.h>
#include <math.h>

#define BATCH 8
#define NPTS  4096

typedef __attribute__((ext_vector_type(16))) _Float16 v16h;
typedef __attribute__((ext_vector_type(8)))  _Float16 v8h;
typedef __attribute__((ext_vector_type(8)))  float    v8f;

#define TM 64
#define TN 64
#define TK 32

// ---------------------------------------------------------------------------
// f32 (rowsIn x K) -> f16 (rowsOut x Kp), zero-padded K tail and zero rows
// beyond rowsIn. Kp % 32 == 0. 8 contiguous halves per thread -> 16B stores.
// ---------------------------------------------------------------------------
__global__ void pn2_cvt_f16(const float* __restrict__ src, _Float16* __restrict__ dst,
                            int rowsIn, int rowsOut, int K, int Kp)
{
    long long t8 = ((long long)blockIdx.x * blockDim.x + threadIdx.x) * 8;
    long long total = (long long)rowsOut * Kp;
    if (t8 >= total) return;
    int c = (int)(t8 % Kp);                 // multiple of 8
    long long r = t8 / Kp;
    v8h tmp = {};
    if (r < rowsIn) {
        const float* s = src + (size_t)r * K;
        #pragma unroll
        for (int e = 0; e < 8; ++e) {
            int cc = c + e;
            int ck = (cc < K) ? cc : (K - 1);   // clamped, in-bounds
            float v = s[ck];
            tmp[e] = (_Float16)((cc < K) ? v : 0.0f);
        }
    }
    *(v8h*)&dst[t8] = tmp;
}

// ---------------------------------------------------------------------------
// Fused GEMM + BatchNorm (+ ReLU):  Y[M,N] = act( (X[M,Kp] @ W[Np,Kp]^T)*s + t )
// X, W pre-converted f16. K zero-padded to Kp (mult of 32); W rows zero-padded
// to Np (mult of 64). REQUIRES M % 64 == 0 (true for every layer here), so the
// hot loop is completely guard-free: 2x global b128 + 2x LDS b128 per thread,
// 6x ds_load_b128 fragments, 2x v_wmma_f32_16x16x32_f16 per wave per K-step.
// Block: 256 threads = 8 waves, tile 64x64, wave slab 16x32.
// ---------------------------------------------------------------------------
__global__ __launch_bounds__(256) void pn2_gemm_bn(
    const _Float16* __restrict__ X, const _Float16* __restrict__ W,
    const float* __restrict__ Bb, const float* __restrict__ Gg,
    const float* __restrict__ Be, const float* __restrict__ Rm,
    const float* __restrict__ Rv, float* __restrict__ Y,
    int M, int Kp, int N, int relu)
{
    __shared__ _Float16 Xs[TM][TK + 8];
    __shared__ _Float16 Ws[TN][TK + 8];

    const int tid  = threadIdx.x;
    const int lane = tid & 31;
    const int wid  = tid >> 5;      // 0..7
    const int wrow = wid & 3;       // 16-row slab of the 64-row tile
    const int wcol = wid >> 2;      // 32-col slab of the 64-col tile

    const int m0 = blockIdx.y * TM;
    const int n0 = blockIdx.x * TN;

    v8f acc0 = {}, acc1 = {};

    // staging map: thread -> 8 contiguous halves of a 64x32 tile (16B aligned)
    const int xr = (tid * 8) >> 5;   // row 0..63
    const int xc = (tid * 8) & 31;   // col 0,8,16,24

    const _Float16* xsrc = X + (size_t)(m0 + xr) * Kp + xc;
    const _Float16* wsrc = W + (size_t)(n0 + xr) * Kp + xc;

    const int ktiles = Kp / TK;
    for (int kt = 0; kt < ktiles; ++kt) {
        const int k0 = kt * TK;

        // ---- stage tiles: pure b128 global->LDS movement, no guards ----
        *(v8h*)&Xs[xr][xc] = *(const v8h*)(xsrc + k0);
        *(v8h*)&Ws[xr][xc] = *(const v8h*)(wsrc + k0);
        if (kt + 1 < ktiles) {
            __builtin_prefetch(xsrc + k0 + TK, 0, 0);   // global_prefetch_b8
        }
        __syncthreads();

        // ---- fragments (wave32): contiguous k order, same for A and B ----
        // A 16x32 f16: lane l -> row M=l&15; elems 0..15 -> K=(l>>4)*16+j
        v16h afrag, bfrag0, bfrag1;
        const int mloc = lane & 15;
        const int half = lane >> 4;
        #pragma unroll
        for (int j = 0; j < 16; ++j) {
            int kl = half * 16 + j;
            afrag[j] = Xs[wrow * 16 + mloc][kl];
        }
        // B 32x16 f16: lane l -> col N=l&15, K=(l>>4)*16 + j
        const int kb = half * 16;
        #pragma unroll
        for (int j = 0; j < 16; ++j) {
            bfrag0[j] = Ws[wcol * 32 + mloc][kb + j];
            bfrag1[j] = Ws[wcol * 32 + 16 + mloc][kb + j];
        }

        acc0 = __builtin_amdgcn_wmma_f32_16x16x32_f16(
            false, afrag, false, bfrag0, (short)0, acc0, false, false);
        acc1 = __builtin_amdgcn_wmma_f32_16x16x32_f16(
            false, afrag, false, bfrag1, (short)0, acc1, false, false);

        __syncthreads();
    }

    // ---- epilogue: C/D layout: VGPR v, lanes0-15 -> M=v, lanes16-31 -> M=v+8
    // M % 64 == 0, so only the padded-N columns need a store guard.
    const int nloc  = lane & 15;
    const int mbase = m0 + wrow * 16 + (lane >> 4) * 8;
    #pragma unroll
    for (int p = 0; p < 2; ++p) {
        const int n = n0 + wcol * 32 + p * 16 + nloc;
        if (n < N) {
            float s = Gg[n] * rsqrtf(Rv[n] + 1e-5f);
            float t = (Bb[n] - Rm[n]) * s + Be[n];
            #pragma unroll
            for (int v = 0; v < 8; ++v) {
                float y = (p == 0 ? acc0[v] : acc1[v]) * s + t;
                if (relu) y = fmaxf(y, 0.0f);
                Y[(size_t)(mbase + v) * N + n] = y;
            }
        }
    }
}

// ---------------------------------------------------------------------------
// (B,3,N) -> (B,N,3)
// ---------------------------------------------------------------------------
__global__ void pn2_transpose_in(const float* __restrict__ xyz, float* __restrict__ pts)
{
    int t = blockIdx.x * blockDim.x + threadIdx.x;
    int total = BATCH * NPTS * 3;
    if (t >= total) return;
    int b = t / (NPTS * 3); int rem = t % (NPTS * 3);
    int n = rem / 3; int d = rem % 3;
    pts[t] = xyz[((size_t)b * 3 + d) * NPTS + n];
}

// ---------------------------------------------------------------------------
// Farthest point sampling: one workgroup per batch, points cached in LDS.
// ---------------------------------------------------------------------------
__global__ __launch_bounds__(256) void pn2_fps(const float* __restrict__ pts,
                                               int* __restrict__ outIdx,
                                               int P, int npoint)
{
    __shared__ float sx[4096], sy[4096], sz[4096];
    __shared__ float rmax[256];
    __shared__ int   ridx[256];

    const int b   = blockIdx.x;
    const int tid = threadIdx.x;
    const float* base = pts + (size_t)b * P * 3;

    for (int i = tid; i < P; i += 256) {
        sx[i] = base[i * 3 + 0];
        sy[i] = base[i * 3 + 1];
        sz[i] = base[i * 3 + 2];
    }
    float dloc[16];
    const int npt = (P + 255) >> 8;
    for (int q = 0; q < npt; ++q) dloc[q] = 1e10f;
    __syncthreads();

    int far = 0;
    if (tid == 0) outIdx[b * npoint] = 0;

    for (int it = 1; it < npoint; ++it) {
        float fx = sx[far], fy = sy[far], fz = sz[far];
        float bm = -1.0f; int bi = 0x7fffffff;
        for (int q = 0; q < npt; ++q) {
            int i = q * 256 + tid;
            if (i < P) {
                float dx = sx[i] - fx, dy = sy[i] - fy, dz = sz[i] - fz;
                float d  = dx * dx + dy * dy + dz * dz;
                float nd = fminf(dloc[q], d);
                dloc[q] = nd;
                if (nd > bm) { bm = nd; bi = i; }
            }
        }
        rmax[tid] = bm; ridx[tid] = bi;
        __syncthreads();
        for (int s = 128; s > 0; s >>= 1) {
            if (tid < s) {
                float o = rmax[tid + s]; int oi = ridx[tid + s];
                if (o > rmax[tid] || (o == rmax[tid] && oi < ridx[tid])) {
                    rmax[tid] = o; ridx[tid] = oi;
                }
            }
            __syncthreads();
        }
        far = ridx[0];
        if (tid == 0) outIdx[b * npoint + it] = far;
        __syncthreads();
    }
}

// ---------------------------------------------------------------------------
// Gather sampled coordinates: dst[b,s,:] = src[b, idx[b,s], :]
// ---------------------------------------------------------------------------
__global__ void pn2_gather3(const float* __restrict__ src, const int* __restrict__ idx,
                            float* __restrict__ dst, int P, int S)
{
    int t = blockIdx.x * blockDim.x + threadIdx.x;
    int total = BATCH * S * 3;
    if (t >= total) return;
    int b = t / (S * 3); int rem = t % (S * 3);
    int s = rem / 3; int d = rem % 3;
    dst[t] = src[((size_t)b * P + idx[b * S + s]) * 3 + d];
}

// ---------------------------------------------------------------------------
// Ball query: ascending-index fill of <=Ksm neighbors within radius.
// ---------------------------------------------------------------------------
__global__ void pn2_ballq(const float* __restrict__ xyz, const float* __restrict__ nxyz,
                          int* __restrict__ gi, int P, int S, int Ksm, float r2)
{
    int t = blockIdx.x * blockDim.x + threadIdx.x;
    int total = BATCH * S;
    if (t >= total) return;
    int b = t / S;
    const float* c = nxyz + (size_t)t * 3;
    float cx = c[0], cy = c[1], cz = c[2];
    const float* base = xyz + (size_t)b * P * 3;
    int* g = gi + (size_t)t * Ksm;
    int cnt = 0;
    for (int j = 0; j < P && cnt < Ksm; ++j) {
        float dx = base[j * 3 + 0] - cx;
        float dy = base[j * 3 + 1] - cy;
        float dz = base[j * 3 + 2] - cz;
        if (dx * dx + dy * dy + dz * dz <= r2) g[cnt++] = j;
    }
    int f = (cnt > 0) ? g[0] : 0;
    for (; cnt < Ksm; ++cnt) g[cnt] = f;
}

// ---------------------------------------------------------------------------
// Build grouped features: G[row, 0:Cf] = feats[gidx], G[row, Cf:Cf+3] = xyz[gidx]-center
// ---------------------------------------------------------------------------
__global__ void pn2_group(const float* __restrict__ xyz, const float* __restrict__ feats,
                          const float* __restrict__ nxyz, const int* __restrict__ gi,
                          float* __restrict__ G, int P, int S, int Ksm, int Cf)
{
    const int Ct = Cf + 3;
    long long t = (long long)blockIdx.x * blockDim.x + threadIdx.x;
    long long total = (long long)BATCH * S * Ksm * Ct;
    if (t >= total) return;
    int c = (int)(t % Ct);
    long long row = t / Ct;
    int b = (int)(row / ((long long)S * Ksm));
    long long rem = row % ((long long)S * Ksm);
    int s = (int)(rem / Ksm);
    int gidx = gi[row];
    if (c < Cf) {
        G[t] = feats[((size_t)b * P + gidx) * Cf + c];
    } else {
        int d = c - Cf;
        G[t] = xyz[((size_t)b * P + gidx) * 3 + d] - nxyz[((size_t)b * S + s) * 3 + d];
    }
}

// ---------------------------------------------------------------------------
// Max over nsample, scattered into channel-concatenated destination.
// ---------------------------------------------------------------------------
__global__ void pn2_maxpool(const float* __restrict__ Yl, float* __restrict__ dst,
                            int BS, int Ksm, int C, int Ctot, int coff)
{
    int t = blockIdx.x * blockDim.x + threadIdx.x;
    int total = BS * C;
    if (t >= total) return;
    int bs = t / C, c = t % C;
    float m = -1e30f;
    for (int j = 0; j < Ksm; ++j)
        m = fmaxf(m, Yl[((size_t)bs * Ksm + j) * C + c]);
    dst[(size_t)bs * Ctot + coff + c] = m;
}

// ---------------------------------------------------------------------------
// sa3 group-all input: [xyz(3), feats(Cf)] per row
// ---------------------------------------------------------------------------
__global__ void pn2_concat_xyzfeat(const float* __restrict__ xyz, const float* __restrict__ feats,
                                   float* __restrict__ G, int rows, int Cf)
{
    const int Ct = 3 + Cf;
    long long t = (long long)blockIdx.x * blockDim.x + threadIdx.x;
    long long total = (long long)rows * Ct;
    if (t >= total) return;
    int c = (int)(t % Ct);
    long long r = t / Ct;
    G[t] = (c < 3) ? xyz[r * 3 + c] : feats[(size_t)r * Cf + (c - 3)];
}

// ---------------------------------------------------------------------------
// fp3 input (S==1 broadcast): [p1 row (C1), p3[b] (C2)]
// ---------------------------------------------------------------------------
__global__ void pn2_fp3_build(const float* __restrict__ p1, const float* __restrict__ p3,
                              float* __restrict__ G, int S1, int C1, int C2)
{
    const int Ct = C1 + C2;
    long long t = (long long)blockIdx.x * blockDim.x + threadIdx.x;
    long long total = (long long)BATCH * S1 * Ct;
    if (t >= total) return;
    int c = (int)(t % Ct);
    long long r = t / Ct;
    int b = (int)(r / S1);
    G[t] = (c < C1) ? p1[(size_t)r * C1 + c] : p3[(size_t)b * C2 + (c - C1)];
}

// ---------------------------------------------------------------------------
// 3-NN inverse-distance interpolation + concat: G = [p1 row, sum_i w_i p2[i]]
// dup=1: p1 is a 3-channel array duplicated to C1=6 (fp1's concat(xyz,pts)).
// ---------------------------------------------------------------------------
__global__ void pn2_fp_interp(const float* __restrict__ xyz1, const float* __restrict__ xyz2,
                              const float* __restrict__ p1, const float* __restrict__ p2,
                              float* __restrict__ G, int N1, int S2, int C1, int C2, int dup)
{
    int t = blockIdx.x * blockDim.x + threadIdx.x;
    int total = BATCH * N1;
    if (t >= total) return;
    int b = t / N1;
    const float* q = xyz1 + (size_t)t * 3;
    float qx = q[0], qy = q[1], qz = q[2];
    const float* base = xyz2 + (size_t)b * S2 * 3;
    float d0 = 1e30f, d1 = 1e30f, d2 = 1e30f;
    int i0 = 0, i1 = 0, i2 = 0;
    for (int j = 0; j < S2; ++j) {
        float dx = base[j * 3 + 0] - qx;
        float dy = base[j * 3 + 1] - qy;
        float dz = base[j * 3 + 2] - qz;
        float d = dx * dx + dy * dy + dz * dz;
        if (d < d0)      { d2 = d1; i2 = i1; d1 = d0; i1 = i0; d0 = d; i0 = j; }
        else if (d < d1) { d2 = d1; i2 = i1; d1 = d;  i1 = j; }
        else if (d < d2) { d2 = d;  i2 = j; }
    }
    float r0 = 1.0f / (d0 + 1e-8f);
    float r1 = 1.0f / (d1 + 1e-8f);
    float r2 = 1.0f / (d2 + 1e-8f);
    float inv = 1.0f / (r0 + r1 + r2);
    float w0 = r0 * inv, w1 = r1 * inv, w2 = r2 * inv;

    float* g = G + (size_t)t * (C1 + C2);
    if (dup) { for (int c = 0; c < C1; ++c) g[c] = p1[(size_t)t * 3 + (c % 3)]; }
    else     { for (int c = 0; c < C1; ++c) g[c] = p1[(size_t)t * C1 + c]; }
    const float* a0 = p2 + ((size_t)b * S2 + i0) * C2;
    const float* a1 = p2 + ((size_t)b * S2 + i1) * C2;
    const float* a2 = p2 + ((size_t)b * S2 + i2) * C2;
    for (int c = 0; c < C2; ++c)
        g[C1 + c] = w0 * a0[c] + w1 * a1[c] + w2 * a2[c];
}

// ---------------------------------------------------------------------------
// (B,N,C) -> (B,C,N)
// ---------------------------------------------------------------------------
__global__ void pn2_transpose_out(const float* __restrict__ Yh, float* __restrict__ out,
                                  int N, int C)
{
    long long t = (long long)blockIdx.x * blockDim.x + threadIdx.x;
    long long total = (long long)BATCH * C * N;
    if (t >= total) return;
    int b = (int)(t / ((long long)C * N));
    long long rem = t % ((long long)C * N);
    int c = (int)(rem / N);
    int n = (int)(rem % N);
    out[t] = Yh[((size_t)b * N + n) * C + c];
}

// ===========================================================================
// Host orchestration
// ===========================================================================
struct BlockP {
    const float *w, *b, *g, *be, *rm, *rv;
    int cin, cout;
};

extern "C" void kernel_launch(void* const* d_in, const int* in_sizes, int n_in,
                              void* d_out, int out_size, void* d_ws, size_t ws_size,
                              hipStream_t stream)
{
    (void)in_sizes; (void)n_in; (void)out_size; (void)ws_size;

    const float* xyz = (const float*)d_in[0];
    float* out = (float*)d_out;

    // --- parameter walker: dict order is w,b,gamma,beta,rm,rv per block ---
    int pi = 1;
    auto nextb = [&](int cin, int cout) -> BlockP {
        BlockP p;
        p.w  = (const float*)d_in[pi + 0];
        p.b  = (const float*)d_in[pi + 1];
        p.g  = (const float*)d_in[pi + 2];
        p.be = (const float*)d_in[pi + 3];
        p.rm = (const float*)d_in[pi + 4];
        p.rv = (const float*)d_in[pi + 5];
        pi += 6;
        p.cin = cin; p.cout = cout;
        return p;
    };

    BlockP sa1[3][3];
    const int sa1w[3][4] = {{6,32,32,64},{6,64,64,128},{6,64,96,128}};
    for (int c = 0; c < 3; ++c)
        for (int l = 0; l < 3; ++l) sa1[c][l] = nextb(sa1w[c][l], sa1w[c][l+1]);

    BlockP sa2[2][3];
    const int sa2w[2][4] = {{323,128,128,256},{323,128,196,256}};
    for (int c = 0; c < 2; ++c)
        for (int l = 0; l < 3; ++l) sa2[c][l] = nextb(sa2w[c][l], sa2w[c][l+1]);

    BlockP sa3[3];
    const int sa3w[4] = {515,256,512,1024};
    for (int l = 0; l < 3; ++l) sa3[l] = nextb(sa3w[l], sa3w[l+1]);

    BlockP fp3[2]; fp3[0] = nextb(1536,256); fp3[1] = nextb(256,256);
    BlockP fp2[2]; fp2[0] = nextb(576,256);  fp2[1] = nextb(256,128);
    BlockP fp1[2]; fp1[0] = nextb(134,128);  fp1[1] = nextb(128,128);
    BlockP head = nextb(128,512);

    // --- workspace bump allocator (byte-based) ---
    char* wp = (char*)d_ws;
    auto allocb = [&](size_t bytes) -> void* {
        void* p = (void*)wp;
        wp += ((bytes + 255) & ~(size_t)255);
        return p;
    };
    auto alloc = [&](size_t elems) -> float* { return (float*)allocb(elems * 4); };

    float* pts    = alloc((size_t)BATCH * NPTS * 3);       // l0 xyz & feats
    int*   fidx1  = (int*)alloc((size_t)BATCH * 512);
    float* nxyz1  = alloc((size_t)BATCH * 512 * 3);        // l1 xyz
    float* l1p    = alloc((size_t)BATCH * 512 * 320);      // l1 points (sa1)
    int*   fidx2  = (int*)alloc((size_t)BATCH * 128);
    float* nxyz2  = alloc((size_t)BATCH * 128 * 3);        // l2 xyz
    float* l2p1   = alloc((size_t)BATCH * 128 * 512);      // l2 points (sa2)
    float* l3p    = alloc((size_t)BATCH * 1024);           // l3 points (sa3)
    float* l2p2   = alloc((size_t)BATCH * 128 * 256);      // l2 points (fp3)
    float* l1p2   = alloc((size_t)BATCH * 512 * 128);      // l1 points (fp2)
    float* l0out  = alloc((size_t)BATCH * NPTS * 128);     // l0 points (fp1)
    int*   gi     = (int*)alloc((size_t)BATCH * 512 * 128);
    float* bufG   = alloc((size_t)42336256);               // max grouped input
    float* bufA   = alloc((size_t)67108864);               // ping
    float* bufB   = alloc((size_t)67108864);               // pong
    _Float16* bufX16 = (_Float16*)allocb((size_t)67108864 * 2);  // padded f16 X
    _Float16* bufW16 = (_Float16*)allocb((size_t)1572864 * 2);   // padded f16 W

    auto cdiv = [](long long a, long long b) -> unsigned {
        return (unsigned)((a + b - 1) / b);
    };

    // All M values here are multiples of 64 (required by pn2_gemm_bn).
    auto run_gemm = [&](const float* Xp, const BlockP& bp, float* Yp, int M, int relu) {
        const int Kp = (bp.cin  + 31) & ~31;
        const int Np = (bp.cout + 63) & ~63;
        pn2_cvt_f16<<<cdiv((long long)M * Kp / 8, 256), 256, 0, stream>>>(
            Xp, bufX16, M, M, bp.cin, Kp);
        pn2_cvt_f16<<<cdiv((long long)Np * Kp / 8, 256), 256, 0, stream>>>(
            bp.w, bufW16, bp.cout, Np, bp.cin, Kp);
        dim3 grid(Np / TN, cdiv(M, TM));
        pn2_gemm_bn<<<grid, dim3(256), 0, stream>>>(
            bufX16, bufW16, bp.b, bp.g, bp.be, bp.rm, bp.rv, Yp, M, Kp, bp.cout, relu);
    };

    auto run_chain = [&](const float* Xin, const BlockP* blks, int nl, int M,
                         float* finalDst, int reluLast) -> const float* {
        const float* cur = Xin;
        for (int i = 0; i < nl; ++i) {
            float* outp = (i == nl - 1 && finalDst) ? finalDst
                        : ((cur == bufA) ? bufB : bufA);
            run_gemm(cur, blks[i], outp, M, (i == nl - 1) ? reluLast : 1);
            cur = outp;
        }
        return cur;
    };

    // ---------- l0 ----------
    pn2_transpose_in<<<cdiv((long long)BATCH * NPTS * 3, 256), 256, 0, stream>>>(xyz, pts);

    // ---------- sa1 (npoint=512, radii .1/.2/.4, nsample 32/64/128) ----------
    pn2_fps<<<BATCH, 256, 0, stream>>>(pts, fidx1, NPTS, 512);
    pn2_gather3<<<cdiv((long long)BATCH * 512 * 3, 256), 256, 0, stream>>>(pts, fidx1, nxyz1, NPTS, 512);
    {
        const int   ns[3]   = {32, 64, 128};
        const float rad[3]  = {0.1f, 0.2f, 0.4f};
        const int   coff[3] = {0, 64, 192};
        const int   co[3]   = {64, 128, 128};
        for (int sc = 0; sc < 3; ++sc) {
            pn2_ballq<<<cdiv((long long)BATCH * 512, 256), 256, 0, stream>>>(
                pts, nxyz1, gi, NPTS, 512, ns[sc], rad[sc] * rad[sc]);
            pn2_group<<<cdiv((long long)BATCH * 512 * ns[sc] * 6, 256), 256, 0, stream>>>(
                pts, pts, nxyz1, gi, bufG, NPTS, 512, ns[sc], 3);
            const float* Yl = run_chain(bufG, sa1[sc], 3, BATCH * 512 * ns[sc], nullptr, 1);
            pn2_maxpool<<<cdiv((long long)BATCH * 512 * co[sc], 256), 256, 0, stream>>>(
                Yl, l1p, BATCH * 512, ns[sc], co[sc], 320, coff[sc]);
        }
    }

    // ---------- sa2 (npoint=128, radii .4/.8, nsample 64/128) ----------
    pn2_fps<<<BATCH, 256, 0, stream>>>(nxyz1, fidx2, 512, 128);
    pn2_gather3<<<cdiv((long long)BATCH * 128 * 3, 256), 256, 0, stream>>>(nxyz1, fidx2, nxyz2, 512, 128);
    {
        const int   ns[2]   = {64, 128};
        const float rad[2]  = {0.4f, 0.8f};
        const int   coff[2] = {0, 256};
        for (int sc = 0; sc < 2; ++sc) {
            pn2_ballq<<<cdiv((long long)BATCH * 128, 256), 256, 0, stream>>>(
                nxyz1, nxyz2, gi, 512, 128, ns[sc], rad[sc] * rad[sc]);
            pn2_group<<<cdiv((long long)BATCH * 128 * ns[sc] * 323, 256), 256, 0, stream>>>(
                nxyz1, l1p, nxyz2, gi, bufG, 512, 128, ns[sc], 320);
            const float* Yl = run_chain(bufG, sa2[sc], 3, BATCH * 128 * ns[sc], nullptr, 1);
            pn2_maxpool<<<cdiv((long long)BATCH * 128 * 256, 256), 256, 0, stream>>>(
                Yl, l2p1, BATCH * 128, ns[sc], 256, 512, coff[sc]);
        }
    }

    // ---------- sa3 (group all: 128 points, Cin=515 -> 1024) ----------
    pn2_concat_xyzfeat<<<cdiv((long long)BATCH * 128 * 515, 256), 256, 0, stream>>>(
        nxyz2, l2p1, bufG, BATCH * 128, 512);
    {
        const float* Y3 = run_chain(bufG, sa3, 3, BATCH * 128, nullptr, 1);
        pn2_maxpool<<<cdiv((long long)BATCH * 1024, 256), 256, 0, stream>>>(
            Y3, l3p, BATCH, 128, 1024, 1024, 0);
    }

    // ---------- fp3 (broadcast l3 -> 128 pts): [l2p1(512), l3(1024)] -> 256 ----------
    pn2_fp3_build<<<cdiv((long long)BATCH * 128 * 1536, 256), 256, 0, stream>>>(
        l2p1, l3p, bufG, 128, 512, 1024);
    run_chain(bufG, fp3, 2, BATCH * 128, l2p2, 1);

    // ---------- fp2 (512 pts <- 128 pts): [l1p(320), interp(256)] -> 128 ----------
    pn2_fp_interp<<<cdiv((long long)BATCH * 512, 256), 256, 0, stream>>>(
        nxyz1, nxyz2, l1p, l2p2, bufG, 512, 128, 320, 256, 0);
    run_chain(bufG, fp2, 2, BATCH * 512, l1p2, 1);

    // ---------- fp1 (4096 pts <- 512 pts): [concat(xyz,pts)=6, interp(128)] -> 128 ----------
    pn2_fp_interp<<<cdiv((long long)BATCH * NPTS, 256), 256, 0, stream>>>(
        pts, nxyz1, pts, l1p2, bufG, NPTS, 512, 6, 128, 1);
    run_chain(bufG, fp1, 2, BATCH * NPTS, l0out, 1);

    // ---------- head (128 -> 512, no relu) + transpose to (B,512,N) ----------
    run_gemm(l0out, head, bufA, BATCH * NPTS, 0);
    pn2_transpose_out<<<cdiv((long long)BATCH * 512 * NPTS, 256), 256, 0, stream>>>(
        bufA, out, NPTS, 512);
}
